// Network_40742059770286
// MI455X (gfx1250) — matrix operand order repaired and checked by reading, not compiled, not run on hardware
//
#include <hip/hip_runtime.h>
#include <hip/hip_bf16.h>
#include <math.h>

#define BB   32
#define NN   64
#define TTOK 512
#define FEAT 768
#define HH   256

typedef __attribute__((ext_vector_type(16))) _Float16 v16h;
typedef __attribute__((ext_vector_type(8)))  _Float16 v8h;
typedef __attribute__((ext_vector_type(8)))  float    v8f;

// ---------------------------------------------------------------------------
// WMMA fragment conventions (per CDNA5 ISA 7.12.2):
// A (16x32 f16, LDS row-major, lda halves): lane L: row=L&15, kb=(L>=16)?8:0;
//   elems 0..7 = K kb..kb+7 ; elems 8..15 = K kb+16..kb+23 (two b128 LDS loads)
// B (32x16 f16, packed 512-half blocks): lane L: col=L&15, K=(L>=16?8:0)+(e&7)+(e>=8?16:0), same order as A.
// D (16x16 f32): VGPR v: row = v + (L>=16?8:0), col = L&15.
// ---------------------------------------------------------------------------
__device__ __forceinline__ v16h load_fragA(const _Float16* __restrict__ p)
{
    v8h lo = *(const v8h*)(p);
    v8h hi = *(const v8h*)(p + 16);
    return __builtin_shufflevector(lo, hi, 0,1,2,3,4,5,6,7,8,9,10,11,12,13,14,15);
}

template<int NK>
__device__ __forceinline__ v8f wmma_rowpanel(const _Float16* __restrict__ ldsA, int lda,
                                             const _Float16* __restrict__ Wp,
                                             int ntile, int lane, v8f acc)
{
    const int row = lane & 15;
    const int kb  = (lane >> 4) << 3;
    const _Float16* arow = ldsA + row * lda + kb;
    const _Float16* bp   = Wp + (size_t)ntile * NK * 512 + lane * 16;
#pragma unroll
    for (int kt = 0; kt < NK; ++kt) {
        v16h a = load_fragA(arow + kt * 32);
        v16h b = *(const v16h*)(bp + (size_t)kt * 512);
        acc = __builtin_amdgcn_wmma_f32_16x16x32_f16(false, a, false, b, (short)0, acc,
                                                     false, false);
    }
    return acc;
}

// ---------------------------------------------------------------------------
// Pack a row-major fp32 weight (K x N) into WMMA-B f16 fragment blocks.
// block = ntile*(K/32) + ktile ; within block: lane*16 + e
// ---------------------------------------------------------------------------
__global__ __launch_bounds__(32) void k_pack_b(const float* __restrict__ W,
                                               _Float16* __restrict__ out,
                                               int K, int N)
{
    const int lane  = threadIdx.x;
    const int block = blockIdx.x;
    const int nK    = K >> 5;
    const int ntile = block / nK;
    const int kt    = block % nK;
    const int n     = ntile * 16 + (lane & 15);
    const int kb    = (lane >> 4) << 3;
    _Float16* o = out + (size_t)block * 512 + lane * 16;
#pragma unroll
    for (int e = 0; e < 16; ++e) {
        int k = kt * 32 + kb + (e & 7) + ((e >> 3) << 4);
        o[e] = (_Float16)W[(size_t)k * N + n];
    }
}

// ---------------------------------------------------------------------------
// K2: gather clause rows + r = doc @ W_red + b_red  -> r_f16 (2048 x 256)
// ---------------------------------------------------------------------------
__global__ __launch_bounds__(256) void k_gather_red(const float* __restrict__ hidden,
                                                    const int* __restrict__ clause,
                                                    const float* __restrict__ b_red,
                                                    const _Float16* __restrict__ Wred_p,
                                                    _Float16* __restrict__ r_f16)
{
    __shared__ _Float16 ldsA[16 * (FEAT + 8)];
    const int tid  = threadIdx.x;
    const int lane = tid & 31;
    const int wave = tid >> 5;
    const int row0 = blockIdx.x * 16;

    for (int idx = tid; idx < 16 * FEAT; idx += 256) {
        int r = idx / FEAT, k = idx - r * FEAT;
        int grow = row0 + r;
        int b = grow >> 6, n = grow & 63;
        int t = clause[b * NN + n];
        ldsA[r * (FEAT + 8) + k] = (_Float16)hidden[((size_t)(b * TTOK + t)) * FEAT + k];
    }
    __syncthreads();

#pragma unroll
    for (int tt = 0; tt < 2; ++tt) {                   // 16 N-tiles / 8 waves
        int ntile = wave * 2 + tt;
        v8f acc = {};
        acc = wmma_rowpanel<FEAT / 32>(ldsA, FEAT + 8, Wred_p, ntile, lane, acc);
        int col   = ntile * 16 + (lane & 15);
        int rbase = (lane >> 4) << 3;
        float bias = b_red[col];
#pragma unroll
        for (int v = 0; v < 8; ++v) {
            int r = v + rbase;
            r_f16[(size_t)(row0 + r) * HH + col] = (_Float16)(acc[v] + bias);
        }
    }
}

// ---------------------------------------------------------------------------
// K3: Acat = r @ W_cat[:H]  ;  Bcat = r @ W_cat[H:]   (fp32, 2048 x 256 each)
// ---------------------------------------------------------------------------
__global__ __launch_bounds__(256) void k_cat(const _Float16* __restrict__ r_f16,
                                             const _Float16* __restrict__ Wp_top,
                                             const _Float16* __restrict__ Wp_bot,
                                             float* __restrict__ Acat,
                                             float* __restrict__ Bcat)
{
    __shared__ _Float16 ldsA[16 * (HH + 8)];
    const int tid  = threadIdx.x;
    const int lane = tid & 31;
    const int wave = tid >> 5;
    const int row0 = blockIdx.x * 16;
    const _Float16* Wp = blockIdx.y ? Wp_bot : Wp_top;
    float* Out         = blockIdx.y ? Bcat : Acat;

    for (int idx = tid; idx < 16 * HH; idx += 256) {
        int r = idx >> 8, k = idx & 255;
        ldsA[r * (HH + 8) + k] = r_f16[(size_t)(row0 + r) * HH + k];
    }
    __syncthreads();

#pragma unroll
    for (int tt = 0; tt < 2; ++tt) {
        int ntile = wave * 2 + tt;
        v8f acc = {};
        acc = wmma_rowpanel<HH / 32>(ldsA, HH + 8, Wp, ntile, lane, acc);
        int col   = ntile * 16 + (lane & 15);
        int rbase = (lane >> 4) << 3;
#pragma unroll
        for (int v = 0; v < 8; ++v)
            Out[(size_t)(row0 + v + rbase) * HH + col] = acc[v];
    }
}

// ---------------------------------------------------------------------------
// K4: xg[b,i,j,:] = relu(Acat[b,i] + Bcat[b,j] + b_cat) @ W_x + b_g
//     M-tile = 32 rows: each B fragment feeds two WMMAs (halves L2 B traffic).
//     xg is write-once/read-once and > L2, so store non-temporally to keep
//     L2 free for the recurrent state Tm (which fits in 192 MB L2).
// ---------------------------------------------------------------------------
__global__ __launch_bounds__(256) void k_xg(const float* __restrict__ Acat,
                                            const float* __restrict__ Bcat,
                                            const float* __restrict__ b_cat,
                                            const float* __restrict__ b_g,
                                            const _Float16* __restrict__ Wx_p,
                                            float* __restrict__ xg)
{
    __shared__ _Float16 ldsA[32 * (HH + 8)];
    const int tid  = threadIdx.x;
    const int lane = tid & 31;
    const int wave = tid >> 5;
    const int rowbase = blockIdx.x * 32;           // global row = (b*64+i)*64+j
    const int b  = rowbase >> 12;
    const int i  = (rowbase >> 6) & 63;
    const int j0 = rowbase & 63;                   // 0 or 32
    const float* arow = Acat + (size_t)(b * NN + i) * HH;

    for (int idx = tid; idx < 32 * HH; idx += 256) {
        int r = idx >> 8, k = idx & 255;
        int j = j0 + r;
        float x = arow[k] + Bcat[(size_t)(b * NN + j) * HH + k] + b_cat[k];
        ldsA[r * (HH + 8) + k] = (_Float16)(x > 0.f ? x : 0.f);
    }
    __syncthreads();

    const int rowA = lane & 15;
    const int kb   = (lane >> 4) << 3;
    const _Float16* a0p = ldsA + rowA * (HH + 8) + kb;
    const _Float16* a1p = a0p + 16 * (HH + 8);

#pragma unroll
    for (int tt = 0; tt < 6; ++tt) {               // 48 N-tiles / 8 waves
        int ntile = wave * 6 + tt;
        const _Float16* bp = Wx_p + (size_t)ntile * 8 * 512 + lane * 16;
        v8f acc0 = {}, acc1 = {};
#pragma unroll
        for (int kt = 0; kt < 8; ++kt) {
            v16h bfrag = *(const v16h*)(bp + (size_t)kt * 512);
            v16h a0 = load_fragA(a0p + kt * 32);
            acc0 = __builtin_amdgcn_wmma_f32_16x16x32_f16(false, a0, false, bfrag,
                                                          (short)0, acc0, false, false);
            v16h a1 = load_fragA(a1p + kt * 32);
            acc1 = __builtin_amdgcn_wmma_f32_16x16x32_f16(false, a1, false, bfrag,
                                                          (short)0, acc1, false, false);
        }
        int col   = ntile * 16 + (lane & 15);
        int rbase = (lane >> 4) << 3;
        float bias = b_g[col];
#pragma unroll
        for (int v = 0; v < 8; ++v) {
            __builtin_nontemporal_store(acc0[v] + bias,
                &xg[(size_t)(rowbase + v + rbase) * 768 + col]);
            __builtin_nontemporal_store(acc1[v] + bias,
                &xg[(size_t)(rowbase + 16 + v + rbase) * 768 + col]);
        }
    }
}

// ---------------------------------------------------------------------------
// K5: one anti-diagonal of the 2-D GRU scan. Workgroup = 16 (batch,cell) rows.
// Prefetches the NEXT diagonal's xg rows into L2 to shorten the serial chain.
// ---------------------------------------------------------------------------
__global__ __launch_bounds__(256) void k_diag(const float* __restrict__ xg,
                                              const _Float16* __restrict__ Uzr_p,
                                              const _Float16* __restrict__ Uh_p,
                                              const int* __restrict__ ymask,
                                              float* __restrict__ Tm, int d)
{
    __shared__ _Float16 spH[16 * (HH + 8)];   // s_prev f16 (GEMM1 A)
    __shared__ float    spF[16 * (HH + 8)];   // s_prev f32
    __shared__ float    zF [16 * (HH + 8)];   // z gate
    __shared__ _Float16 rsH[16 * (HH + 8)];   // r * s_prev (GEMM2 A)

    const int tid  = threadIdx.x;
    const int lane = tid & 31;
    const int wave = tid >> 5;

    const int i_lo = (d > 63) ? (d - 63) : 0;
    const int cell = blockIdx.x >> 1;
    const int b0   = (blockIdx.x & 1) * 16;
    const int i    = i_lo + cell;
    const int j    = d - i;

    // prefetch next diagonal's xg rows (cells (i,j+1) and (i+1,j)) into L2
    // 768 floats per row -> 12 x 256B lines ; 16 batches x 2 neighbors = 384 ops
    for (int idx = tid; idx < 384; idx += 256) {
        int line  = idx % 12;
        int bb    = (idx / 12) & 15;
        int which = idx / 192;
        int pi = which ? (i + 1) : i;
        int pj = which ? j : (j + 1);
        if (pi < NN && pj < NN) {
            size_t g = ((size_t)((b0 + bb) * NN + pi)) * NN + pj;
            __builtin_prefetch(&xg[g * 768 + line * 64], 0, 1);
        }
    }

    // phase 0: s_prev = 0.5*(s_left + s_up)
    for (int idx = tid; idx < 16 * HH; idx += 256) {
        int r = idx >> 8, k = idx & 255;
        int b = b0 + r;
        float sl = (j > 0) ? Tm[(((size_t)(b * NN + i)) * NN + (j - 1)) * HH + k] : 0.f;
        float su = (i > 0) ? Tm[(((size_t)(b * NN + (i - 1))) * NN + j) * HH + k] : 0.f;
        float sp = 0.5f * (sl + su);
        spF[r * (HH + 8) + k] = sp;
        spH[r * (HH + 8) + k] = (_Float16)sp;
    }
    __syncthreads();

    // phase 1: zr = sigmoid(xg[:, :512] + s_prev @ Us_zr)
#pragma unroll
    for (int tt = 0; tt < 4; ++tt) {               // 32 N-tiles (N=512) / 8 waves
        int ntile = wave * 4 + tt;
        v8f acc = {};
        acc = wmma_rowpanel<HH / 32>(spH, HH + 8, Uzr_p, ntile, lane, acc);
        int col   = ntile * 16 + (lane & 15);
        int rbase = (lane >> 4) << 3;
#pragma unroll
        for (int v = 0; v < 8; ++v) {
            int r = v + rbase;
            int b = b0 + r;
            size_t g = ((size_t)(b * NN + i)) * NN + j;
            float pre = acc[v] + __builtin_nontemporal_load(&xg[g * 768 + col]);
            float sg  = 1.f / (1.f + __expf(-pre));
            if (col < HH) {
                zF[r * (HH + 8) + col] = sg;
            } else {
                int rc = col - HH;
                rsH[r * (HH + 8) + rc] = (_Float16)(sg * spF[r * (HH + 8) + rc]);
            }
        }
    }
    __syncthreads();

    // phase 2: h = tanh(xg[:, 512:] + (r*s_prev) @ Us_h) ; s = ((1-z)sp + z h)*m
#pragma unroll
    for (int tt = 0; tt < 2; ++tt) {               // 16 N-tiles / 8 waves
        int ntile = wave * 2 + tt;
        v8f acc = {};
        acc = wmma_rowpanel<HH / 32>(rsH, HH + 8, Uh_p, ntile, lane, acc);
        int col   = ntile * 16 + (lane & 15);
        int rbase = (lane >> 4) << 3;
#pragma unroll
        for (int v = 0; v < 8; ++v) {
            int r = v + rbase;
            int b = b0 + r;
            size_t g = ((size_t)(b * NN + i)) * NN + j;
            float h  = tanhf(acc[v] + __builtin_nontemporal_load(&xg[g * 768 + 2 * HH + col]));
            float z  = zF[r * (HH + 8) + col];
            float sp = spF[r * (HH + 8) + col];
            float m  = (ymask[b * NN + i] & ymask[b * NN + j]) ? 1.f : 0.f;
            Tm[g * HH + col] = ((1.f - z) * sp + z * h) * m;
        }
    }
}

// ---------------------------------------------------------------------------
// K6: couples_pred = Tm @ W_t + b_t   (131072 x 2) ; one wave per row
// ---------------------------------------------------------------------------
__global__ __launch_bounds__(256) void k_couples(const float* __restrict__ Tm,
                                                 const float* __restrict__ W_t,
                                                 const float* __restrict__ b_t,
                                                 float* __restrict__ out)
{
    const int lane = threadIdx.x & 31;
    const int wave = threadIdx.x >> 5;
    const size_t row = (size_t)blockIdx.x * 8 + wave;
    const float* trow = Tm + row * HH;
    float a0 = 0.f, a1 = 0.f;
    for (int k = lane; k < HH; k += 32) {
        float t = trow[k];
        a0 += t * W_t[k * 2 + 0];
        a1 += t * W_t[k * 2 + 1];
    }
    for (int off = 16; off > 0; off >>= 1) {
        a0 += __shfl_down(a0, off, 32);
        a1 += __shfl_down(a1, off, 32);
    }
    if (lane == 0) {
        out[row * 2 + 0] = a0 + b_t[0];
        out[row * 2 + 1] = a1 + b_t[1];
    }
}

// ---------------------------------------------------------------------------
// K7: pred_e / pred_c from diagonal rows Tm[b,n,n,:]
// ---------------------------------------------------------------------------
__global__ __launch_bounds__(256) void k_diagpred(const float* __restrict__ Tm,
                                                  const float* __restrict__ W_e,
                                                  const float* __restrict__ b_e,
                                                  const float* __restrict__ W_c,
                                                  const float* __restrict__ b_c,
                                                  float* __restrict__ out_e,
                                                  float* __restrict__ out_c)
{
    const int lane = threadIdx.x & 31;
    const int wave = threadIdx.x >> 5;
    const int row  = blockIdx.x * 8 + wave;        // 2048 rows
    const int b = row >> 6, n = row & 63;
    const float* trow = Tm + (((size_t)(b * NN + n)) * NN + n) * HH;
    float e0 = 0.f, e1 = 0.f, c0 = 0.f, c1 = 0.f;
    for (int k = lane; k < HH; k += 32) {
        float t = trow[k];
        e0 += t * W_e[k * 2 + 0];
        e1 += t * W_e[k * 2 + 1];
        c0 += t * W_c[k * 2 + 0];
        c1 += t * W_c[k * 2 + 1];
    }
    for (int off = 16; off > 0; off >>= 1) {
        e0 += __shfl_down(e0, off, 32);
        e1 += __shfl_down(e1, off, 32);
        c0 += __shfl_down(c0, off, 32);
        c1 += __shfl_down(c1, off, 32);
    }
    if (lane == 0) {
        out_e[row * 2 + 0] = e0 + b_e[0];
        out_e[row * 2 + 1] = e1 + b_e[1];
        out_c[row * 2 + 0] = c0 + b_c[0];
        out_c[row * 2 + 1] = c1 + b_c[1];
    }
}

// ---------------------------------------------------------------------------
extern "C" void kernel_launch(void* const* d_in, const int* in_sizes, int n_in,
                              void* d_out, int out_size, void* d_ws, size_t ws_size,
                              hipStream_t stream)
{
    (void)in_sizes; (void)n_in; (void)out_size; (void)ws_size;

    const float* hidden = (const float*)d_in[0];
    const float* W_red  = (const float*)d_in[1];
    const float* b_red  = (const float*)d_in[2];
    const float* W_cat  = (const float*)d_in[3];
    const float* b_cat  = (const float*)d_in[4];
    const float* W_x    = (const float*)d_in[5];
    const float* Us_zr  = (const float*)d_in[6];
    const float* Us_h   = (const float*)d_in[7];
    const float* b_g    = (const float*)d_in[8];
    const float* W_e    = (const float*)d_in[9];
    const float* b_e    = (const float*)d_in[10];
    const float* W_c    = (const float*)d_in[11];
    const float* b_c    = (const float*)d_in[12];
    const float* W_t    = (const float*)d_in[13];
    const float* b_t    = (const float*)d_in[14];
    const int*   clause = (const int*)d_in[15];
    const int*   ymask  = (const int*)d_in[16];
    float* out = (float*)d_out;

    char* ws = (char*)d_ws;
    size_t off = 0;
    auto carve = [&](size_t bytes) -> void* {
        void* p = ws + off;
        off += (bytes + 255) & ~(size_t)255;
        return p;
    };
    _Float16* Wred_p  = (_Float16*)carve((size_t)FEAT * HH * 2);
    _Float16* Wtop_p  = (_Float16*)carve((size_t)HH * HH * 2);
    _Float16* Wbot_p  = (_Float16*)carve((size_t)HH * HH * 2);
    _Float16* Wx_p    = (_Float16*)carve((size_t)HH * 768 * 2);
    _Float16* Uzr_p   = (_Float16*)carve((size_t)HH * 512 * 2);
    _Float16* Uh_p    = (_Float16*)carve((size_t)HH * HH * 2);
    _Float16* r_f16   = (_Float16*)carve((size_t)BB * NN * HH * 2);
    float*    Acat    = (float*)carve((size_t)BB * NN * HH * 4);
    float*    Bcat    = (float*)carve((size_t)BB * NN * HH * 4);
    float*    xg      = (float*)carve((size_t)BB * NN * NN * 768 * 4);
    float*    Tm      = (float*)carve((size_t)BB * NN * NN * HH * 4);

    // pack weights into WMMA-B fragment layout (f16)
    k_pack_b<<<(HH / 16) * (FEAT / 32), 32, 0, stream>>>(W_red, Wred_p, FEAT, HH);
    k_pack_b<<<(HH / 16) * (HH / 32),   32, 0, stream>>>(W_cat, Wtop_p, HH, HH);
    k_pack_b<<<(HH / 16) * (HH / 32),   32, 0, stream>>>(W_cat + (size_t)HH * HH, Wbot_p, HH, HH);
    k_pack_b<<<(768 / 16) * (HH / 32),  32, 0, stream>>>(W_x, Wx_p, HH, 768);
    k_pack_b<<<(512 / 16) * (HH / 32),  32, 0, stream>>>(Us_zr, Uzr_p, HH, 512);
    k_pack_b<<<(HH / 16) * (HH / 32),   32, 0, stream>>>(Us_h, Uh_p, HH, HH);

    // gather + reduction GEMM
    k_gather_red<<<(BB * NN) / 16, 256, 0, stream>>>(hidden, clause, b_red, Wred_p, r_f16);

    // pairwise-concat GEMM halves
    k_cat<<<dim3((BB * NN) / 16, 2), 256, 0, stream>>>(r_f16, Wtop_p, Wbot_p, Acat, Bcat);

    // fused relu(Acat_i + Bcat_j + b_cat) @ W_x + b_g  (M-tile 32)
    k_xg<<<(BB * NN * NN) / 32, 256, 0, stream>>>(Acat, Bcat, b_cat, b_g, Wx_p, xg);

    // 2-D wavefront GRU scan, one launch per anti-diagonal
    for (int d = 0; d < 2 * NN - 1; ++d) {
        int i_lo = (d > NN - 1) ? (d - (NN - 1)) : 0;
        int i_hi = (d < NN - 1) ? d : (NN - 1);
        int ncells = i_hi - i_lo + 1;
        k_diag<<<ncells * 2, 256, 0, stream>>>(xg, Uzr_p, Uh_p, ymask, Tm, d);
    }

    // outputs
    float* out_couples = out;                                  // 32*64*64*2
    float* out_e       = out + (size_t)BB * NN * NN * 2;       // 32*64*2
    float* out_c       = out_e + (size_t)BB * NN * 2;          // 32*64*2
    k_couples<<<(BB * NN * NN) / 8, 256, 0, stream>>>(Tm, W_t, b_t, out_couples);
    k_diagpred<<<(BB * NN) / 8, 256, 0, stream>>>(Tm, W_e, b_e, W_c, b_c, out_e, out_c);
}
